// Encoder_63316407878001
// MI455X (gfx1250) — matrix operand order repaired and checked by reading, not compiled, not run on hardware
//
#include <hip/hip_runtime.h>

typedef __bf16 bf16_t;
typedef __attribute__((ext_vector_type(16))) __bf16 bf16x16;
typedef __attribute__((ext_vector_type(8)))  __bf16 bf16x8;
typedef __attribute__((ext_vector_type(8)))  float  f32x8;

#define DEV static __device__ __forceinline__

union BFrag { bf16x16 v; bf16x8 h[2]; };

DEV bf16x8 ld8(const bf16_t* p){ return *(const bf16x8*)p; }

// A-operand 16x32 bf16, row-major source (stride lda elements).
// Per ISA 7.12.2: lanes 0-15 row=lane hold K {0..7, 16..23}; lanes 16-31 hold K {8..15, 24..31}.
DEV bf16x16 load_a_rm(const bf16_t* A, int lda, int m0, int k0, int lane){
  int hl = lane >> 4, r = lane & 15;
  const bf16_t* p = A + (size_t)(m0 + r) * lda + k0 + hl * 8;
  BFrag f; f.h[0] = ld8(p); f.h[1] = ld8(p + 16);
  return f.v;
}

// B-operand 32x16 (KxN) from a weight stored [N][K] row-major:
// lane n = column; lanes 0-15 hold K {0..7, 16..23}; lanes 16-31 hold K {8..15, 24..31}.
DEV bf16x16 load_b_nm(const bf16_t* W, int ldw, int n0, int k0, int lane){
  int hl = lane >> 4, r = lane & 15;
  const bf16_t* p = W + (size_t)(n0 + r) * ldw + k0 + hl * 8;
  BFrag f; f.h[0] = ld8(p); f.h[1] = ld8(p + 16);
  return f.v;
}

// A-operand from virtual concat [h | ctx] along K (each 512 wide, k0 multiple of 32).
DEV bf16x16 load_a_cat(const bf16_t* hb, const bf16_t* cx, int m0, int k0, int lane){
  const bf16_t* base = (k0 < 512) ? hb : cx;
  int k = (k0 < 512) ? k0 : (k0 - 512);
  return load_a_rm(base, 512, m0, k, lane);
}

DEV f32x8 wmma_bf16(bf16x16 a, bf16x16 b, f32x8 c){
  // (neg_a, A, neg_b, B, c_mod, C, reuse_a, reuse_b)
  return __builtin_amdgcn_wmma_f32_16x16x32_bf16(false, a, false, b, (short)0, c, false, false);
}

DEV float fsig(float x){ return 1.0f / (1.0f + __expf(-x)); }
DEV float ftanh(float x){
  x = fminf(fmaxf(x, -15.0f), 15.0f);
  float e = __expf(2.0f * x);
  return (e - 1.0f) / (e + 1.0f);
}

// ---------------------------------------------------------------- conversions
__global__ void k_cvt_bf16(const float* __restrict__ s, bf16_t* __restrict__ d, int n){
  int i = blockIdx.x * blockDim.x + threadIdx.x;
  if (i < n) d[i] = (bf16_t)s[i];
}

// -------------------------------------------------- xz = x @ Wih^T + b  (both dirs)
// A = x_bf16 [8192][768]; B = Wih [1024][768]; out Z [dir][t][b][1024] (backward time-reversed)
// Software-pipelined: fragments for k-step kk+1 are in flight while WMMAs for kk execute.
__global__ __launch_bounds__(256) void k_xz_gemm(
    const bf16_t* __restrict__ xbf, const bf16_t* __restrict__ wihbf,
    const float* __restrict__ bias_f, const float* __restrict__ bias_b,
    float* __restrict__ Z)
{
  const int dir  = blockIdx.y;
  const int lane = threadIdx.x & 31;
  const int hl = lane >> 4, rl = lane & 15;
  const int wid = blockIdx.x * 8 + (threadIdx.x >> 5);
  const int m0 = (wid >> 4) * 16;
  const int n0 = (wid & 15) * 64;
  const bf16_t* W = wihbf + (size_t)dir * 1024 * 768;
  const float* bias = dir ? bias_b : bias_f;
  float* Zd = Z + (size_t)dir * 256 * 32 * 1024;

  f32x8 acc[4];
  #pragma unroll
  for (int g = 0; g < 4; ++g){
    float bv = bias[n0 + g * 16 + rl];
    #pragma unroll
    for (int v = 0; v < 8; ++v) acc[g][v] = bv;
  }

  bf16x16 a0 = load_a_rm(xbf, 768, m0, 0, lane);
  bf16x16 b0[4], b1[4];
  #pragma unroll
  for (int g = 0; g < 4; ++g) b0[g] = load_b_nm(W, 768, n0 + g * 16, 0, lane);

  for (int kk = 0; kk < 24; kk += 2){
    // prefetch kk+1, compute kk
    int k1 = (kk + 1) * 32;
    bf16x16 a1 = load_a_rm(xbf, 768, m0, k1, lane);
    #pragma unroll
    for (int g = 0; g < 4; ++g) b1[g] = load_b_nm(W, 768, n0 + g * 16, k1, lane);
    #pragma unroll
    for (int g = 0; g < 4; ++g) acc[g] = wmma_bf16(a0, b0[g], acc[g]);
    // prefetch kk+2 (clamped), compute kk+1
    int k2 = (kk + 2 < 24) ? (kk + 2) * 32 : k1;
    a0 = load_a_rm(xbf, 768, m0, k2, lane);
    #pragma unroll
    for (int g = 0; g < 4; ++g) b0[g] = load_b_nm(W, 768, n0 + g * 16, k2, lane);
    #pragma unroll
    for (int g = 0; g < 4; ++g) acc[g] = wmma_bf16(a1, b1[g], acc[g]);
  }

  #pragma unroll
  for (int v = 0; v < 8; ++v){
    int gr = m0 + v + hl * 8;            // row = b*256 + t
    int bb = gr >> 8, tt = gr & 255;
    int zt = dir ? (255 - tt) : tt;
    float* zrow = Zd + ((size_t)zt * 32 + bb) * 1024;
    #pragma unroll
    for (int g = 0; g < 4; ++g) zrow[n0 + g * 16 + rl] = acc[g][v];
  }
}

// -------------------------------------------------- persistent LSTM recurrence
// One block (32 waves) per direction. Wave w owns m-tile (w>>4) and hidden-col tile j=w&15
// for ALL FOUR gates, so the gate nonlinearity is local to the wave. h lives in LDS (bf16),
// c lives in accumulator-layout registers. B operand (Whh from L2) is ping-pong prefetched.
__global__ __launch_bounds__(1024) void k_lstm(
    const float* __restrict__ Z, const bf16_t* __restrict__ whhbf,
    float* __restrict__ hf32, bf16_t* __restrict__ hbf, bf16_t* __restrict__ hT)
{
  __shared__ bf16_t hsh[32 * 256];
  const int dir  = blockIdx.x;
  const int tid  = threadIdx.x;
  const int lane = tid & 31;
  const int hl = lane >> 4, rl = lane & 15;
  const int w  = tid >> 5;
  const int m0 = (w >> 4) * 16;
  const int j  = w & 15;
  const float* Zd = Z + (size_t)dir * 256 * 32 * 1024;
  const bf16_t* Whh = whhbf + (size_t)dir * 1024 * 256;

  #pragma unroll
  for (int i = 0; i < 8; ++i) hsh[tid + i * 1024] = (bf16_t)0.0f;
  f32x8 creg;
  #pragma unroll
  for (int v = 0; v < 8; ++v) creg[v] = 0.0f;
  __syncthreads();

  for (int t = 0; t < 256; ++t){
    const float* Zt = Zd + (size_t)t * 32 * 1024;
    f32x8 acc[4];
    #pragma unroll
    for (int g = 0; g < 4; ++g){
      #pragma unroll
      for (int v = 0; v < 8; ++v)
        acc[g][v] = Zt[(size_t)(m0 + v + hl * 8) * 1024 + g * 256 + j * 16 + rl];
    }

    bf16x16 b0[4], b1[4];
    #pragma unroll
    for (int g = 0; g < 4; ++g) b0[g] = load_b_nm(Whh, 256, g * 256 + j * 16, 0, lane);
    #pragma unroll
    for (int kk = 0; kk < 8; kk += 2){
      int k1 = (kk + 1) * 32;
      #pragma unroll
      for (int g = 0; g < 4; ++g) b1[g] = load_b_nm(Whh, 256, g * 256 + j * 16, k1, lane);
      bf16x16 a = load_a_rm(hsh, 256, m0, kk * 32, lane);      // ds_load fragments
      #pragma unroll
      for (int g = 0; g < 4; ++g) acc[g] = wmma_bf16(a, b0[g], acc[g]);
      int k2 = (kk + 2 < 8) ? (kk + 2) * 32 : k1;
      #pragma unroll
      for (int g = 0; g < 4; ++g) b0[g] = load_b_nm(Whh, 256, g * 256 + j * 16, k2, lane);
      a = load_a_rm(hsh, 256, m0, k1, lane);
      #pragma unroll
      for (int g = 0; g < 4; ++g) acc[g] = wmma_bf16(a, b1[g], acc[g]);
    }

    __syncthreads();                     // all reads of h(t-1) done
    const int at = dir ? (255 - t) : t;
    #pragma unroll
    for (int v = 0; v < 8; ++v){
      float iv = acc[0][v], fv = acc[1][v], gv = acc[2][v], ov = acc[3][v];
      float cn = fsig(fv) * creg[v] + fsig(iv) * ftanh(gv);
      float hn = fsig(ov) * ftanh(cn);
      creg[v] = cn;
      int bb = m0 + v + hl * 8;
      int col = j * 16 + rl;
      hsh[bb * 256 + col] = (bf16_t)hn;
      size_t o = ((size_t)bb * 256 + at) * 512 + dir * 256 + col;
      hf32[o] = hn;
      hbf[o]  = (bf16_t)hn;
      hT[((size_t)bb * 512 + dir * 256 + col) * 256 + at] = (bf16_t)hn;
    }
    __syncthreads();                     // h(t) visible for next step
  }
}

// -------------------------------------------------- sq/sk: 10-dim score projections
__global__ __launch_bounds__(256) void k_proj_qk(
    const float* __restrict__ h, const float* __restrict__ W1,
    float* __restrict__ sq, float* __restrict__ sk)
{
  const int wid  = blockIdx.x * 8 + (threadIdx.x >> 5);   // bt row
  const int lane = threadIdx.x & 31;
  const float* hr = h + (size_t)wid * 512;
  for (int x = 0; x < 10; ++x){
    float s1 = 0.f, s2 = 0.f;
    for (int d = lane; d < 512; d += 32){
      float hv = hr[d];
      s1 += hv * W1[x * 1024 + d];
      s2 += hv * W1[x * 1024 + 512 + d];
    }
    #pragma unroll
    for (int o = 16; o > 0; o >>= 1){
      s1 += __shfl_xor(s1, o, 32);
      s2 += __shfl_xor(s2, o, 32);
    }
    if (lane == 0){ sq[wid * 10 + x] = s1; sk[wid * 10 + x] = s2; }
  }
}

// -------------------------------------------------- additive logits + softmax -> bf16 weights
__global__ __launch_bounds__(256) void k_attn_softmax(
    const float* __restrict__ sq, const float* __restrict__ sk,
    const float* __restrict__ W2, const float* __restrict__ mask,
    bf16_t* __restrict__ wbuf)
{
  const int wid  = blockIdx.x * 8 + (threadIdx.x >> 5);   // b*256+q
  const int lane = threadIdx.x & 31;
  const int b = wid >> 8;
  float sqv[10], w2v[10];
  #pragma unroll
  for (int x = 0; x < 10; ++x){ sqv[x] = sq[wid * 10 + x]; w2v[x] = W2[x]; }
  float a[8];
  #pragma unroll
  for (int ii = 0; ii < 8; ++ii){
    int k = lane + ii * 32;
    const float* skr = sk + (size_t)(b * 256 + k) * 10;
    float s = 0.f;
    #pragma unroll
    for (int x = 0; x < 10; ++x) s += ftanh(sqv[x] + skr[x]) * w2v[x];
    if (mask[b * 256 + k] == 0.0f) s = -1e30f;
    a[ii] = s;
  }
  float mx = a[0];
  #pragma unroll
  for (int ii = 1; ii < 8; ++ii) mx = fmaxf(mx, a[ii]);
  #pragma unroll
  for (int o = 16; o > 0; o >>= 1) mx = fmaxf(mx, __shfl_xor(mx, o, 32));
  float sum = 0.f;
  #pragma unroll
  for (int ii = 0; ii < 8; ++ii){ a[ii] = __expf(a[ii] - mx); sum += a[ii]; }
  #pragma unroll
  for (int o = 16; o > 0; o >>= 1) sum += __shfl_xor(sum, o, 32);
  float inv = 1.0f / sum;
  #pragma unroll
  for (int ii = 0; ii < 8; ++ii)
    wbuf[(size_t)wid * 256 + lane + ii * 32] = (bf16_t)(a[ii] * inv);
}

// -------------------------------------------------- context = w @ h  (per batch, WMMA)
__global__ __launch_bounds__(256) void k_context(
    const bf16_t* __restrict__ wbuf,   // [32][256][256]
    const bf16_t* __restrict__ hT,     // [32][512 d][256 k]
    bf16_t* __restrict__ ctx)          // [8192][512]
{
  const int lane = threadIdx.x & 31;
  const int hl = lane >> 4, rl = lane & 15;
  const int job = blockIdx.x * 8 + (threadIdx.x >> 5);   // 4096 jobs
  const int b  = job >> 7;
  const int m0 = ((job >> 3) & 15) * 16;
  const int n0 = (job & 7) * 64;
  const bf16_t* A  = wbuf + (size_t)b * 256 * 256;
  const bf16_t* Bm = hT   + (size_t)b * 512 * 256;
  f32x8 acc[4];
  #pragma unroll
  for (int g = 0; g < 4; ++g)
    #pragma unroll
    for (int v = 0; v < 8; ++v) acc[g][v] = 0.0f;

  bf16x16 a0 = load_a_rm(A, 256, m0, 0, lane);
  bf16x16 b0[4], b1[4];
  #pragma unroll
  for (int g = 0; g < 4; ++g) b0[g] = load_b_nm(Bm, 256, n0 + g * 16, 0, lane);
  #pragma unroll
  for (int kk = 0; kk < 8; kk += 2){
    int k1 = (kk + 1) * 32;
    bf16x16 a1 = load_a_rm(A, 256, m0, k1, lane);
    #pragma unroll
    for (int g = 0; g < 4; ++g) b1[g] = load_b_nm(Bm, 256, n0 + g * 16, k1, lane);
    #pragma unroll
    for (int g = 0; g < 4; ++g) acc[g] = wmma_bf16(a0, b0[g], acc[g]);
    int k2 = (kk + 2 < 8) ? (kk + 2) * 32 : k1;
    a0 = load_a_rm(A, 256, m0, k2, lane);
    #pragma unroll
    for (int g = 0; g < 4; ++g) b0[g] = load_b_nm(Bm, 256, n0 + g * 16, k2, lane);
    #pragma unroll
    for (int g = 0; g < 4; ++g) acc[g] = wmma_bf16(a1, b1[g], acc[g]);
  }

  #pragma unroll
  for (int v = 0; v < 8; ++v){
    size_t row = (size_t)b * 256 + m0 + v + hl * 8;
    #pragma unroll
    for (int g = 0; g < 4; ++g)
      ctx[row * 512 + n0 + g * 16 + rl] = (bf16_t)acc[g][v];
  }
}

// -------------------------------------------------- y = [h|ctx] @ W3^T + b3
__global__ __launch_bounds__(256) void k_output(
    const bf16_t* __restrict__ hbf, const bf16_t* __restrict__ ctx,
    const bf16_t* __restrict__ w3bf, const float* __restrict__ b3,
    float* __restrict__ y)
{
  const int lane = threadIdx.x & 31;
  const int hl = lane >> 4, rl = lane & 15;
  const int job = blockIdx.x * 8 + (threadIdx.x >> 5);   // 4096 jobs
  const int m0 = (job >> 3) * 16;
  const int n0 = (job & 7) * 64;
  f32x8 acc[4];
  #pragma unroll
  for (int g = 0; g < 4; ++g){
    float bv = b3[n0 + g * 16 + rl];
    #pragma unroll
    for (int v = 0; v < 8; ++v) acc[g][v] = bv;
  }

  bf16x16 a0 = load_a_cat(hbf, ctx, m0, 0, lane);
  bf16x16 b0[4], b1[4];
  #pragma unroll
  for (int g = 0; g < 4; ++g) b0[g] = load_b_nm(w3bf, 1024, n0 + g * 16, 0, lane);

  for (int kk = 0; kk < 32; kk += 2){
    int k1 = (kk + 1) * 32;
    bf16x16 a1 = load_a_cat(hbf, ctx, m0, k1, lane);
    #pragma unroll
    for (int g = 0; g < 4; ++g) b1[g] = load_b_nm(w3bf, 1024, n0 + g * 16, k1, lane);
    #pragma unroll
    for (int g = 0; g < 4; ++g) acc[g] = wmma_bf16(a0, b0[g], acc[g]);
    int k2 = (kk + 2 < 32) ? (kk + 2) * 32 : k1;
    a0 = load_a_cat(hbf, ctx, m0, k2, lane);
    #pragma unroll
    for (int g = 0; g < 4; ++g) b0[g] = load_b_nm(w3bf, 1024, n0 + g * 16, k2, lane);
    #pragma unroll
    for (int g = 0; g < 4; ++g) acc[g] = wmma_bf16(a1, b1[g], acc[g]);
  }

  #pragma unroll
  for (int v = 0; v < 8; ++v){
    size_t row = (size_t)(m0 + v + hl * 8);
    #pragma unroll
    for (int g = 0; g < 4; ++g)
      y[row * 512 + n0 + g * 16 + rl] = acc[g][v];
  }
}

// ============================================================================
extern "C" void kernel_launch(void* const* d_in, const int* in_sizes, int n_in,
                              void* d_out, int out_size, void* d_ws, size_t ws_size,
                              hipStream_t stream){
  const float* x     = (const float*)d_in[0];
  const float* mask  = (const float*)d_in[1];
  const float* Wih_f = (const float*)d_in[2];
  const float* Whh_f = (const float*)d_in[3];
  const float* b_f   = (const float*)d_in[4];
  const float* Wih_b = (const float*)d_in[5];
  const float* Whh_b = (const float*)d_in[6];
  const float* b_b   = (const float*)d_in[7];
  const float* W1    = (const float*)d_in[8];
  const float* W2    = (const float*)d_in[9];
  const float* W3    = (const float*)d_in[10];
  const float* b3    = (const float*)d_in[11];
  float* y = (float*)d_out;
  (void)in_sizes; (void)n_in; (void)out_size; (void)ws_size;

  char* ws = (char*)d_ws;
  size_t off = 0;
  auto alloc = [&](size_t bytes)->char*{
    char* p = ws + off;
    off += (bytes + 255) & ~(size_t)255;
    return p;
  };
  bf16_t* xbf   = (bf16_t*)alloc((size_t)8192*768*2);
  bf16_t* wihbf = (bf16_t*)alloc((size_t)2*1024*768*2);
  bf16_t* whhbf = (bf16_t*)alloc((size_t)2*1024*256*2);
  bf16_t* w3bf  = (bf16_t*)alloc((size_t)512*1024*2);
  float*  Z     = (float*) alloc((size_t)2*256*32*1024*4);
  float*  hf32  = (float*) alloc((size_t)8192*512*4);
  bf16_t* hbf   = (bf16_t*)alloc((size_t)8192*512*2);
  bf16_t* hT    = (bf16_t*)alloc((size_t)32*512*256*2);
  float*  sq    = (float*) alloc((size_t)8192*10*4);
  float*  sk    = (float*) alloc((size_t)8192*10*4);
  bf16_t* wbuf  = (bf16_t*)alloc((size_t)32*256*256*2);
  bf16_t* ctx   = (bf16_t*)alloc((size_t)8192*512*2);

  auto cvt = [&](const float* s, bf16_t* d, int n){
    k_cvt_bf16<<<(n + 255)/256, 256, 0, stream>>>(s, d, n);
  };
  cvt(x, xbf, 8192*768);
  cvt(Wih_f, wihbf, 1024*768);
  cvt(Wih_b, wihbf + (size_t)1024*768, 1024*768);
  cvt(Whh_f, whhbf, 1024*256);
  cvt(Whh_b, whhbf + (size_t)1024*256, 1024*256);
  cvt(W3, w3bf, 512*1024);

  k_xz_gemm<<<dim3(1024, 2), 256, 0, stream>>>(xbf, wihbf, b_f, b_b, Z);
  k_lstm<<<2, 1024, 0, stream>>>(Z, whhbf, hf32, hbf, hT);
  k_proj_qk<<<1024, 256, 0, stream>>>(hf32, W1, sq, sk);
  k_attn_softmax<<<1024, 256, 0, stream>>>(sq, sk, W2, mask, wbuf);
  k_context<<<512, 256, 0, stream>>>(wbuf, hT, ctx);
  k_output<<<512, 256, 0, stream>>>(hbf, ctx, w3bf, b3, y);
}
